// MeWNetwork_49804440765133
// MI455X (gfx1250) — matrix operand order repaired and checked by reading, not compiled, not run on hardware
//
#include <hip/hip_runtime.h>
#include <hip/hip_bf16.h>
#include <stdint.h>

#define HDIM 2048
#define G4   (4 * HDIM)
#define N_OPS 8
#define N_NODES 7

typedef __attribute__((ext_vector_type(16))) __bf16 v16bf;
typedef __attribute__((ext_vector_type(8)))  float  v8f;

union Frag32 { v16bf v; float4 f[2]; };

// ---------------------------------------------------------------------------
// async memory->LDS staging (CDNA5 GLOBAL_LOAD_ASYNC_TO_LDS, ASYNCcnt path)
// one instruction per wave moves 32 lanes x 16B = 512B directly into LDS.
// ---------------------------------------------------------------------------
__device__ inline void async_copy_b128(uint32_t lds_off, const void* gptr) {
    asm volatile("global_load_async_to_lds_b128 %0, %1, off"
                 :: "v"(lds_off), "v"((uint64_t)(uintptr_t)gptr)
                 : "memory");
}
__device__ inline void wait_asynccnt0() {
    asm volatile("s_wait_asynccnt 0x0" ::: "memory");
}

// ---------------------------------------------------------------------------
// f32 -> bf16 weight conversion (one pass; halves all later HBM traffic)
// ---------------------------------------------------------------------------
__global__ void cvt_bf16_kernel(const float* __restrict__ src,
                                __bf16* __restrict__ dst, long long n) {
    long long i = (long long)blockIdx.x * blockDim.x + threadIdx.x;
    long long stride = (long long)gridDim.x * blockDim.x;
    for (; i < n; i += stride) dst[i] = (__bf16)src[i];
}

// ---------------------------------------------------------------------------
// init state: h=c=0, anchors[0..1]=0, x = bf16(w_emb[N_OPS]), scalars = 0
// ---------------------------------------------------------------------------
__global__ void init_kernel(const float* __restrict__ w_emb,
                            float* __restrict__ h, float* __restrict__ c,
                            __bf16* __restrict__ hbf, __bf16* __restrict__ xbf,
                            float* __restrict__ anchors, float* __restrict__ scal) {
    int i = blockIdx.x * blockDim.x + threadIdx.x;
    if (i < 2 * HDIM) { h[i] = 0.f; c[i] = 0.f; hbf[i] = (__bf16)0.f; anchors[i] = 0.f; }
    if (i < HDIM) xbf[i] = (__bf16)w_emb[N_OPS * HDIM + i];
    if (i == 0) { scal[0] = 0.f; scal[1] = 0.f; }
}

// ---------------------------------------------------------------------------
// WMMA bf16 GEMV with split-K=2:  y = W1*x1 (+ W2*x2) (+ bias1 + bias2)
// W row-major [nrows x HDIM]. 256-thread block = 8 waves: waves 0-3 do the
// low K-half of tiles 0-3, waves 4-7 the high K-half; partials combined in LDS.
// A fragment = x chunk replicated across all 16 M-rows, so every row of the
// accumulator equals x.B; lanes 0-15 extract from acc[0] (M=0, N=lane).
// B fragment: lane n holds W[out0 + n%16, k0 + (n>=16)*16 .. +15] (contiguous,
// same k order as the A fragment: lanes 0-15 K=0-15, lanes 16-31 K=16-31).
// ---------------------------------------------------------------------------
__global__ __launch_bounds__(256)
void gemv_wmma_kernel(const __bf16* __restrict__ W1, const __bf16* __restrict__ x1,
                      const __bf16* __restrict__ W2, const __bf16* __restrict__ x2,
                      const float* __restrict__ bias1, const float* __restrict__ bias2,
                      float* __restrict__ y, int nrows) {
    __shared__ __align__(16) __bf16 sx1[HDIM];
    __shared__ __align__(16) __bf16 sx2[HDIM];
    __shared__ float sred[4][2][16];

    // stage x vector(s) into LDS via async copies: 256 chunks of 16B each
    {
        const int i = threadIdx.x;                  // blockDim.x == HDIM*2/16
        const uint32_t o1 = (uint32_t)(uintptr_t)sx1 + i * 16;
        async_copy_b128(o1, (const char*)x1 + i * 16);
        if (W2) {
            const uint32_t o2 = (uint32_t)(uintptr_t)sx2 + i * 16;
            async_copy_b128(o2, (const char*)x2 + i * 16);
        }
        wait_asynccnt0();                           // per-wave ASYNCcnt drain
    }
    __syncthreads();

    const int lane = threadIdx.x & 31;
    const int wave = threadIdx.x >> 5;
    const int tid4 = wave & 3;                      // tile slot in block
    const int ksel = wave >> 2;                     // 0: K[0,H/2), 1: K[H/2,H)
    const int tile = blockIdx.x * 4 + tid4;
    const int out0 = tile * 16;
    const int col  = lane & 15;
    const int kh   = lane >> 4;                     // K 0-15 vs 16-31 half of chunk
    const bool alive = (out0 < nrows);
    const int ks = ksel * (HDIM / 2);

    v8f acc = {};
    if (alive) {
        const size_t rowoff = (size_t)(out0 + col) * HDIM + (size_t)kh * 16 + ks;
        {
            const __bf16* brow = W1 + rowoff;
            const __bf16* sx   = sx1 + ks;
            for (int k0 = 0; k0 < HDIM / 2; k0 += 32) {
                if ((k0 & 127) == 0)
                    __builtin_prefetch(brow + k0 + 1024, 0, 0);   // global_prefetch_b8
                Frag32 a, b;
                a.f[0] = *(const float4*)(sx + k0 + kh * 16);
                a.f[1] = *(const float4*)(sx + k0 + kh * 16 + 8);
                const float4* pb = (const float4*)(brow + k0);
                b.f[0] = pb[0]; b.f[1] = pb[1];
                acc = __builtin_amdgcn_wmma_f32_16x16x32_bf16(false, a.v, false, b.v,
                                                              (short)0, acc, false, false);
            }
        }
        if (W2) {
            const __bf16* brow = W2 + rowoff;
            const __bf16* sx   = sx2 + ks;
            for (int k0 = 0; k0 < HDIM / 2; k0 += 32) {
                if ((k0 & 127) == 0)
                    __builtin_prefetch(brow + k0 + 1024, 0, 0);
                Frag32 a, b;
                a.f[0] = *(const float4*)(sx + k0 + kh * 16);
                a.f[1] = *(const float4*)(sx + k0 + kh * 16 + 8);
                const float4* pb = (const float4*)(brow + k0);
                b.f[0] = pb[0]; b.f[1] = pb[1];
                acc = __builtin_amdgcn_wmma_f32_16x16x32_bf16(false, a.v, false, b.v,
                                                              (short)0, acc, false, false);
            }
        }
    }

    if (lane < 16) sred[tid4][ksel][col] = acc[0];
    __syncthreads();

    if (alive && ksel == 0 && lane < 16) {
        float r = sred[tid4][0][col] + sred[tid4][1][col];
        if (bias1) r += bias1[out0 + col];
        if (bias2) r += bias2[out0 + col];
        y[out0 + col] = r;
    }
}

// ---------------------------------------------------------------------------
// LSTM cell nonlinearity: gates g[4H] (i,f,g,o) -> c,h; also bf16 copy of h.
// ---------------------------------------------------------------------------
__global__ void lstm_update_kernel(const float* __restrict__ g, float* __restrict__ c,
                                   float* __restrict__ h, __bf16* __restrict__ hbf) {
    int i = blockIdx.x * blockDim.x + threadIdx.x;
    if (i >= HDIM) return;
    float gi = g[i], gf = g[HDIM + i], gg = g[2 * HDIM + i], go = g[3 * HDIM + i];
    float si = 1.f / (1.f + expf(-gi));
    float sf = 1.f / (1.f + expf(-gf));
    float so = 1.f / (1.f + expf(-go));
    float c2 = sf * c[i] + si * tanhf(gg);
    float h2 = so * tanhf(c2);
    c[i] = c2; h[i] = h2; hbf[i] = (__bf16)h2;
}

// ---------------------------------------------------------------------------
// attention logits: logits[j] = 2.5*tanh( (v . tanh(aw1[j] + hq)) / 5 )
// ---------------------------------------------------------------------------
__global__ void attn_logits_kernel(const float* __restrict__ anchors_w1,
                                   const float* __restrict__ hq,
                                   const float* __restrict__ vattn,
                                   float* __restrict__ logits) {
    int j = blockIdx.x;
    float s = 0.f;
    for (int i = threadIdx.x; i < HDIM; i += blockDim.x)
        s += vattn[i] * tanhf(anchors_w1[j * HDIM + i] + hq[i]);
    __shared__ float red[256];
    red[threadIdx.x] = s;
    __syncthreads();
    for (int st = 128; st > 0; st >>= 1) {
        if (threadIdx.x < st) red[threadIdx.x] += red[threadIdx.x + st];
        __syncthreads();
    }
    if (threadIdx.x == 0) logits[j] = 2.5f * tanhf(red[0] / 5.0f);
}

// ---------------------------------------------------------------------------
// op logits: logits[r] = (2.5/2.5)*tanh( (w_soft[r].h + b[r]) / 5 )
// ---------------------------------------------------------------------------
__global__ void op_logits_kernel(const float* __restrict__ wsoft,
                                 const float* __restrict__ wsoft_b,
                                 const float* __restrict__ h1,
                                 float* __restrict__ logits) {
    int r = blockIdx.x;
    float s = 0.f;
    for (int i = threadIdx.x; i < HDIM; i += blockDim.x)
        s += wsoft[r * HDIM + i] * h1[i];
    __shared__ float red[256];
    red[threadIdx.x] = s;
    __syncthreads();
    for (int st = 128; st > 0; st >>= 1) {
        if (threadIdx.x < st) red[threadIdx.x] += red[threadIdx.x + st];
        __syncthreads();
    }
    if (threadIdx.x == 0) logits[r] = tanhf((red[0] + wsoft_b[r]) / 5.0f);
}

// ---------------------------------------------------------------------------
// categorical sampling (deterministic splitmix64 stand-in for threefry),
// accumulates log_p / entropy, records idx + arc entry.
// ---------------------------------------------------------------------------
__device__ inline float u01_from(uint64_t x) {
    x += 0x9E3779B97F4A7C15ull;
    x = (x ^ (x >> 30)) * 0xBF58476D1CE4E5B9ull;
    x = (x ^ (x >> 27)) * 0x94D049BB133111EBull;
    x = x ^ (x >> 31);
    return (float)(x >> 40) * (1.0f / 16777216.0f);
}

__global__ void sample_kernel(const float* __restrict__ logits, int n, int t,
                              float* __restrict__ scal, int* __restrict__ idx_out,
                              float* __restrict__ arc_out) {
    float mx = -3.4e38f;
    for (int i = 0; i < n; ++i) mx = fmaxf(mx, logits[i]);
    float se = 0.f;
    for (int i = 0; i < n; ++i) se += expf(logits[i] - mx);
    float lse = mx + logf(se);
    float u = u01_from(((uint64_t)42u << 32) | (uint32_t)t);
    int idx = n - 1;
    float cdf = 0.f;
    for (int i = 0; i < n; ++i) {
        cdf += expf(logits[i] - lse);
        if (u < cdf) { idx = i; break; }
    }
    float ent = 0.f;
    for (int i = 0; i < n; ++i) {
        float lp = logits[i] - lse;
        ent -= lp * expf(lp);
    }
    scal[0] += logits[idx] - lse;
    scal[1] += ent;
    *idx_out = idx;
    if (arc_out) *arc_out = (float)idx;
}

// ---------------------------------------------------------------------------
// gathers / copies for next-step input selection
// ---------------------------------------------------------------------------
__global__ void gather_anchor_kernel(const float* __restrict__ anchors,
                                     const int* __restrict__ idxp,
                                     __bf16* __restrict__ xbf) {
    int i = blockIdx.x * blockDim.x + threadIdx.x;
    int idx = *idxp;
    if (i < HDIM) xbf[i] = (__bf16)anchors[idx * HDIM + i];
}

__global__ void gather_emb_kernel(const float* __restrict__ wemb,
                                  const int* __restrict__ idxp, int imm,
                                  __bf16* __restrict__ xbf) {
    int i = blockIdx.x * blockDim.x + threadIdx.x;
    int idx = idxp ? *idxp : imm;
    if (i < HDIM) xbf[i] = (__bf16)wemb[idx * HDIM + i];
}

__global__ void copy_f32_kernel(float* __restrict__ dst, const float* __restrict__ src, int n) {
    int i = blockIdx.x * blockDim.x + threadIdx.x;
    if (i < n) dst[i] = src[i];
}

__global__ void finalize_kernel(const float* __restrict__ scal, float* __restrict__ out) {
    out[4 * N_NODES + 0] = scal[0];
    out[4 * N_NODES + 1] = scal[1];
}

// ---------------------------------------------------------------------------
// host orchestration (all on `stream`, graph-capture safe)
// ---------------------------------------------------------------------------
extern "C" void kernel_launch(void* const* d_in, const int* in_sizes, int n_in,
                              void* d_out, int out_size, void* d_ws, size_t ws_size,
                              hipStream_t stream) {
    (void)in_sizes; (void)n_in; (void)ws_size;
    const float* w_emb    = (const float*)d_in[0];
    const float* emb_attn = (const float*)d_in[1];
    const float* hid_attn = (const float*)d_in[2];
    const float* v_attn   = (const float*)d_in[3];
    const float* w_soft   = (const float*)d_in[4];
    const float* w_soft_b = (const float*)d_in[5];
    const float* w_ih     = (const float*)d_in[6];
    const float* w_hh     = (const float*)d_in[7];
    const float* b_ih     = (const float*)d_in[8];
    const float* b_hh     = (const float*)d_in[9];

    char* base = (char*)d_ws; size_t off = 0;
    auto alloc = [&](size_t bytes) -> void* {
        void* p = base + off; off = (off + bytes + 255) & ~(size_t)255; return p;
    };
    const size_t LW = (size_t)2 * G4 * HDIM;  // elems of [L,4H,H]
    __bf16* wih_bf = (__bf16*)alloc(LW * 2);
    __bf16* whh_bf = (__bf16*)alloc(LW * 2);
    __bf16* ea_bf  = (__bf16*)alloc((size_t)HDIM * HDIM * 2);
    __bf16* ha_bf  = (__bf16*)alloc((size_t)HDIM * HDIM * 2);
    float*  g      = (float*)alloc((size_t)G4 * 4);
    float*  h      = (float*)alloc((size_t)2 * HDIM * 4);
    float*  c      = (float*)alloc((size_t)2 * HDIM * 4);
    __bf16* hbf    = (__bf16*)alloc((size_t)2 * HDIM * 2);
    __bf16* xbf    = (__bf16*)alloc((size_t)HDIM * 2);
    float*  hq     = (float*)alloc((size_t)HDIM * 4);
    float*  anchors    = (float*)alloc((size_t)9 * HDIM * 4);
    float*  anchors_w1 = (float*)alloc((size_t)9 * HDIM * 4);
    float*  logits = (float*)alloc(16 * 4);
    float*  scal   = (float*)alloc(2 * 4);
    int*    idxb   = (int*)alloc(4);

    float* out = (float*)d_out;
    const bool full_out = (out_size >= 4 * N_NODES + 2);

    // one-time bf16 conversion of the streamed weights
    cvt_bf16_kernel<<<2048, 256, 0, stream>>>(w_ih, wih_bf, (long long)LW);
    cvt_bf16_kernel<<<2048, 256, 0, stream>>>(w_hh, whh_bf, (long long)LW);
    cvt_bf16_kernel<<<512, 256, 0, stream>>>(emb_attn, ea_bf, (long long)HDIM * HDIM);
    cvt_bf16_kernel<<<512, 256, 0, stream>>>(hid_attn, ha_bf, (long long)HDIM * HDIM);

    init_kernel<<<(2 * HDIM) / 256, 256, 0, stream>>>(w_emb, h, c, hbf, xbf, anchors, scal);

    auto lstm_step = [&]() {
        for (int l = 0; l < 2; ++l) {
            const __bf16* xin = (l == 0) ? xbf : (hbf + 0);
            gemv_wmma_kernel<<<G4 / 64, 256, 0, stream>>>(
                wih_bf + (size_t)l * G4 * HDIM, xin,
                whh_bf + (size_t)l * G4 * HDIM, hbf + l * HDIM,
                b_ih + l * G4, b_hh + l * G4, g, G4);
            lstm_update_kernel<<<HDIM / 256, 256, 0, stream>>>(g, c + l * HDIM,
                                                               h + l * HDIM, hbf + l * HDIM);
        }
    };

    int t = 0;
    // warm-up: two zero-anchor registrations
    for (int a = 0; a < 2; ++a) {
        lstm_step();
        gemv_wmma_kernel<<<HDIM / 64, 256, 0, stream>>>(
            ea_bf, hbf + HDIM, nullptr, nullptr, nullptr, nullptr,
            anchors_w1 + (size_t)a * HDIM, HDIM);
    }

    for (int node = 0; node < N_NODES; ++node) {
        int n_anch = node + 2;
        for (int rep = 0; rep < 2; ++rep) {           // predecessor-node decisions
            lstm_step();
            gemv_wmma_kernel<<<HDIM / 64, 256, 0, stream>>>(
                ha_bf, hbf + HDIM, nullptr, nullptr, nullptr, nullptr, hq, HDIM);
            attn_logits_kernel<<<n_anch, 256, 0, stream>>>(anchors_w1, hq, v_attn, logits);
            sample_kernel<<<1, 1, 0, stream>>>(logits, n_anch, t, scal, idxb,
                                               full_out ? out + node * 4 + rep * 2 : nullptr);
            ++t;
            gather_anchor_kernel<<<HDIM / 256, 256, 0, stream>>>(anchors, idxb, xbf);
        }
        for (int rep = 0; rep < 2; ++rep) {           // op decisions
            lstm_step();
            op_logits_kernel<<<N_OPS, 256, 0, stream>>>(w_soft, w_soft_b, h + HDIM, logits);
            sample_kernel<<<1, 1, 0, stream>>>(logits, N_OPS, t, scal, idxb,
                                               full_out ? out + node * 4 + 1 + rep * 2 : nullptr);
            ++t;
            gather_emb_kernel<<<HDIM / 256, 256, 0, stream>>>(w_emb, idxb, 0, xbf);
        }
        // anchor registration for this node
        lstm_step();
        gemv_wmma_kernel<<<HDIM / 64, 256, 0, stream>>>(
            ea_bf, hbf + HDIM, nullptr, nullptr, nullptr, nullptr,
            anchors_w1 + (size_t)n_anch * HDIM, HDIM);
        copy_f32_kernel<<<HDIM / 256, 256, 0, stream>>>(anchors + (size_t)n_anch * HDIM,
                                                        h + HDIM, HDIM);
        gather_emb_kernel<<<HDIM / 256, 256, 0, stream>>>(w_emb, nullptr, N_OPS, xbf);
    }
    if (full_out) finalize_kernel<<<1, 1, 0, stream>>>(scal, out);
}
